// Attention_25202868093662
// MI455X (gfx1250) — matrix operand order repaired and checked by reading, not compiled, not run on hardware
//
#include <hip/hip_runtime.h>
#include <cstdint>
#include <cstddef>

// ---------------------------------------------------------------------------
// GLM dual-RoPE attention for MI455X (gfx1250, wave32, WMMA).
// All GEMM-like math uses V_WMMA_F32_16X16X32_BF16 (bf16 in, f32 accum).
// ---------------------------------------------------------------------------

typedef __attribute__((ext_vector_type(16))) __bf16 v16bf;
typedef __attribute__((ext_vector_type(8)))  __bf16 v8bf;
typedef __attribute__((ext_vector_type(8)))  float  v8f;

union BF16x16 { v16bf v; v8bf h[2]; };

// Problem constants (from reference setup_inputs)
#define SEQ   2048
#define DMODEL 2048
#define NHEAD 16
#define HDIM  128
#define D3    (3 * DMODEL)
#define L2E   1.4426950408889634f

// A-fragment (16xK, 16-bit, K=32): lane<16 holds row M=lane, K {0..7,16..23};
// lane>=16 holds row M=lane-16, K {8..15,24..31}.  base = &A[row][kbase].
__device__ __forceinline__ v16bf load_frag_a(const __bf16* base, int lane) {
  const int half = (lane >> 4) << 3;        // 0 or 8
  BF16x16 r;
  r.h[0] = *reinterpret_cast<const v8bf*>(base + half);
  r.h[1] = *reinterpret_cast<const v8bf*>(base + 16 + half);
  return r.v;
}

// B-fragment (Kx16, 16-bit, K=32), column = lane&15:
// lanes 0-15 hold K {0..7,16..23}, lanes 16-31 hold K {8..15,24..31}. base = &Bt[col][kbase].
__device__ __forceinline__ v16bf load_frag_b(const __bf16* base, int lane) {
  const int off = (lane >> 4) << 3;         // 0 or 8
  BF16x16 r;
  r.h[0] = *reinterpret_cast<const v8bf*>(base + off);
  r.h[1] = *reinterpret_cast<const v8bf*>(base + off + 16);
  return r.v;
}

__device__ __forceinline__ v8f wmma_bf(v16bf a, v16bf b, v8f c) {
  return __builtin_amdgcn_wmma_f32_16x16x32_bf16(
      /*neg_a=*/false, a, /*neg_b=*/false, b,
      /*c_mod=*/(short)0, c, /*reuse_a=*/false, /*reuse_b=*/false);
}

// ---------------------------------------------------------------------------
// Elementwise helpers
// ---------------------------------------------------------------------------
__global__ void cast_f32_bf16(const float* __restrict__ src,
                              __bf16* __restrict__ dst, int n) {
  int i = blockIdx.x * blockDim.x + threadIdx.x;
  if (i < n) dst[i] = (__bf16)src[i];
}

// dst[n*K + k] = src[k*N + n]  (transpose + cast; coalesced reads over n)
__global__ void tcast_f32_bf16(const float* __restrict__ src,
                               __bf16* __restrict__ dst, int K, int N) {
  int i = blockIdx.x * blockDim.x + threadIdx.x;
  if (i >= K * N) return;
  int n = i % N, k = i / N;
  dst[(size_t)n * K + k] = (__bf16)src[(size_t)k * N + n];
}

// vt[(h*HDIM+d)*SEQ + s] = qkv[s*D3 + 2*DMODEL + h*HDIM + d]
__global__ void transpose_v(const __bf16* __restrict__ qkv,
                            __bf16* __restrict__ vt) {
  int i = blockIdx.x * blockDim.x + threadIdx.x;
  if (i >= SEQ * DMODEL) return;
  int c = i & (DMODEL - 1);
  int s = i >> 11;
  vt[(size_t)c * SEQ + s] = qkv[(size_t)s * D3 + 2 * DMODEL + c];
}

// GLM dual RoPE, in place on the q and k slabs of qkv (bf16).
// idx bits: j[5] | halfsel[1] | qk[1] | h[4] | s[11]
__global__ void rope_kernel(__bf16* __restrict__ qkv,
                            const int* __restrict__ pos_ids) {
  int i = blockIdx.x * blockDim.x + threadIdx.x;
  if (i >= SEQ * NHEAD * 2 * 2 * 32) return;
  int j    = i & 31;
  int hsel = (i >> 5) & 1;            // 0 -> first 64 dims (pos stream 0)
  int qk   = (i >> 6) & 1;            // 0 -> q, 1 -> k
  int h    = (i >> 7) & 15;
  int s    = i >> 11;
  float pos  = (float)pos_ids[hsel * SEQ + s];          // [1,2,S] flat
  // inv_freq[j] = 10000^(-j/32) = 2^(-j*log2(10000)/32)
  float ang  = pos * exp2f(-(float)j * (13.287712379549449f / 32.0f));
  float c = cosf(ang), sn = sinf(ang);
  __bf16* base = qkv + (size_t)s * D3 + qk * DMODEL + h * HDIM + hsel * 64 + j;
  float x1 = (float)base[0];
  float x2 = (float)base[32];
  base[0]  = (__bf16)(x1 * c - x2 * sn);
  base[32] = (__bf16)(x2 * c + x1 * sn);
}

// ---------------------------------------------------------------------------
// GEMM: out[M,N] = A[M,K](bf16,row-major) x Bt[N,K](bf16) + bias[N]
// One wave -> 64(M) x 64(N): 4 A-frags x 4 B-frags -> 16 independent WMMAs
// per k-step (16 b128 loads/lane), hiding WMMA RAW latency and maximizing
// math per byte from L2.  Block = 4 waves in 2x2 -> 128 x 128 tile.
// ---------------------------------------------------------------------------
template <bool OUT_F32>
__global__ __launch_bounds__(128)
void gemm_bf16(const __bf16* __restrict__ A, const __bf16* __restrict__ Bt,
               const float* __restrict__ bias, void* __restrict__ out,
               int M, int N, int K) {
  const int lane = threadIdx.x & 31;
  const int wave = threadIdx.x >> 5;
  const int lo = lane & 15, hi = lane >> 4;
  const int m0 = blockIdx.y * 128 + (wave >> 1) * 64;
  const int n0 = blockIdx.x * 128 + (wave & 1) * 64;

  const __bf16* aptr[4];
  const __bf16* bptr[4];
#pragma unroll
  for (int t = 0; t < 4; ++t) {
    aptr[t] = A  + (size_t)(m0 + t * 16 + lo) * K;
    bptr[t] = Bt + (size_t)(n0 + t * 16 + lo) * K;
  }

  v8f acc[4][4];
#pragma unroll
  for (int i = 0; i < 4; ++i)
#pragma unroll
    for (int j = 0; j < 4; ++j) acc[i][j] = (v8f){0,0,0,0,0,0,0,0};

  for (int kb = 0; kb < K; kb += 32) {
    v16bf af[4], bfr[4];
#pragma unroll
    for (int t = 0; t < 4; ++t) af[t]  = load_frag_a(aptr[t] + kb, lane);
#pragma unroll
    for (int t = 0; t < 4; ++t) bfr[t] = load_frag_b(bptr[t] + kb, lane);
    if (kb + 128 < K) {
      // pull the k+4 step's lines toward the WGP (global_prefetch_b8)
#pragma unroll
      for (int t = 0; t < 4; ++t) {
        __builtin_prefetch(aptr[t] + kb + 128, 0, 1);
        __builtin_prefetch(bptr[t] + kb + 128, 0, 1);
      }
    }
#pragma unroll
    for (int i = 0; i < 4; ++i)
#pragma unroll
      for (int j = 0; j < 4; ++j)
        acc[i][j] = wmma_bf(af[i], bfr[j], acc[i][j]);
  }

#pragma unroll
  for (int i = 0; i < 4; ++i)
#pragma unroll
    for (int j = 0; j < 4; ++j) {
      int col = n0 + j * 16 + lo;
      float bv = bias[col];
#pragma unroll
      for (int r = 0; r < 8; ++r) {
        int row = m0 + i * 16 + r + hi * 8;       // C/D layout
        float v = acc[i][j][r] + bv;
        if (OUT_F32) ((float*)out)[(size_t)row * N + col] = v;
        else         ((__bf16*)out)[(size_t)row * N + col] = (__bf16)v;
      }
    }
}

// ---------------------------------------------------------------------------
// Flash attention (unscaled, unmasked, softmax over keys).
// One wave per (head, 16-query tile).  Logits computed TRANSPOSED
// (Lt = Ktile x Q^T) so softmaxed P lands exactly in the A-fragment layout
// needed for the P x V WMMA -- no cross-lane data movement for P.
// ---------------------------------------------------------------------------
__global__ __launch_bounds__(128)
void flash_attn(const __bf16* __restrict__ qkv, const __bf16* __restrict__ vt,
                __bf16* __restrict__ ctx) {
  const int lane  = threadIdx.x & 31;
  const int wave  = threadIdx.x >> 5;
  const int qtile = blockIdx.x * 4 + wave;        // 0..127
  const int h     = blockIdx.y;                   // 0..15
  const int lo = lane & 15, hi = lane >> 4;

  // Preload Q^T B-fragments: column = query qtile*16+lo, 4 d-chunks of 32.
  v16bf qf[4];
  {
    const __bf16* qb = qkv + (size_t)(qtile * 16 + lo) * D3 + h * HDIM;
#pragma unroll
    for (int kc = 0; kc < 4; ++kc) qf[kc] = load_frag_b(qb + kc * 32, lane);
  }

  v8f o[8];
#pragma unroll
  for (int t = 0; t < 8; ++t) o[t] = (v8f){0,0,0,0,0,0,0,0};
  float m = -1e30f, l = 0.0f;                     // state for query q = lo

  for (int kt = 0; kt < SEQ; kt += 32) {
    // ---- Lt tiles: rows = keys (kt..kt+31), cols = queries ----
    v8f lt0 = (v8f){0,0,0,0,0,0,0,0};
    v8f lt1 = (v8f){0,0,0,0,0,0,0,0};
#pragma unroll
    for (int kc = 0; kc < 4; ++kc) {
      const __bf16* kb0 =
          qkv + (size_t)(kt + lo) * D3 + DMODEL + h * HDIM + kc * 32;
      v16bf kf0 = load_frag_a(kb0, lane);
      v16bf kf1 = load_frag_a(kb0 + (size_t)16 * D3, lane);
      lt0 = wmma_bf(kf0, qf[kc], lt0);
      lt1 = wmma_bf(kf1, qf[kc], lt1);
    }

    // ---- online softmax over this 32-key slab (per query q = lo) ----
    float tmax = lt0[0];
#pragma unroll
    for (int r = 0; r < 8; ++r) {
      tmax = fmaxf(tmax, lt0[r]);
      tmax = fmaxf(tmax, lt1[r]);
    }
    tmax = fmaxf(tmax, __shfl_xor(tmax, 16));
    float mnew  = fmaxf(m, tmax);
    float scale = exp2f((m - mnew) * L2E);
    float rs = 0.0f;
#pragma unroll
    for (int r = 0; r < 8; ++r) {
      lt0[r] = exp2f((lt0[r] - mnew) * L2E); rs += lt0[r];
      lt1[r] = exp2f((lt1[r] - mnew) * L2E); rs += lt1[r];
    }
    rs += __shfl_xor(rs, 16);
    l = l * scale + rs;
    m = mnew;

    // ---- pack P as A-fragment (16q x 32k): elems 0..7 <- lt0, 8..15 <- lt1
    BF16x16 pf;
#pragma unroll
    for (int r = 0; r < 8; ++r) {
      pf.v[r]     = (__bf16)lt0[r];
      pf.v[r + 8] = (__bf16)lt1[r];
    }

    // ---- rescale O (row q' = r + 8*hi; its scale lives in lane q') ----
    float sc[8];
#pragma unroll
    for (int r = 0; r < 8; ++r) sc[r] = __shfl(scale, r + 8 * hi);
#pragma unroll
    for (int t = 0; t < 8; ++t)
#pragma unroll
      for (int r = 0; r < 8; ++r) o[t][r] *= sc[r];

    // ---- O += P x V  (B-frag from pre-transposed V: contiguous keys) ----
#pragma unroll
    for (int t = 0; t < 8; ++t) {
      const __bf16* vb = vt + (size_t)(h * HDIM + t * 16 + lo) * SEQ + kt;
      v16bf vf = load_frag_b(vb, lane);
      o[t] = wmma_bf(pf.v, vf, o[t]);
    }
  }

  // ---- epilogue: divide rows by l, store ctx[q, h*HDIM + d] ----
  float linv = 1.0f / l;
  float li[8];
#pragma unroll
  for (int r = 0; r < 8; ++r) li[r] = __shfl(linv, r + 8 * hi);
#pragma unroll
  for (int t = 0; t < 8; ++t)
#pragma unroll
    for (int r = 0; r < 8; ++r) {
      int row = qtile * 16 + r + 8 * hi;
      ctx[(size_t)row * DMODEL + h * HDIM + t * 16 + lo] =
          (__bf16)(o[t][r] * li[r]);
    }
}

// ---------------------------------------------------------------------------
// Host-side orchestration
// ---------------------------------------------------------------------------
extern "C" void kernel_launch(void* const* d_in, const int* in_sizes, int n_in,
                              void* d_out, int out_size, void* d_ws, size_t ws_size,
                              hipStream_t stream) {
  const float* x      = (const float*)d_in[0];   // [S, 1, D]
  const float* w_qkv  = (const float*)d_in[1];   // [D, 3D]
  const float* b_qkv  = (const float*)d_in[2];   // [3D]
  const float* w_out  = (const float*)d_in[3];   // [D, D]
  const float* b_out  = (const float*)d_in[4];   // [D]
  const int*   posids = (const int*)d_in[5];     // [1, 2, S]
  float* out = (float*)d_out;                    // [S, 1, D] f32

  // Workspace layout (bf16 unless noted), total 80 MB
  char* ws = (char*)d_ws;
  __bf16* xb    = (__bf16*)(ws);                         //  8 MB  x
  __bf16* wqkvT = (__bf16*)(ws + (8u   << 20));          // 24 MB  [3D, D]
  __bf16* woutT = (__bf16*)(ws + (32u  << 20));          //  8 MB  [D, D]
  __bf16* qkvb  = (__bf16*)(ws + (40u  << 20));          // 24 MB  [S, 3D]
  __bf16* vtb   = (__bf16*)(ws + (64u  << 20));          //  8 MB  [D, S]
  __bf16* ctxb  = (__bf16*)(ws + (72u  << 20));          //  8 MB  [S, D]

  const int TPB = 256;
  // 1) casts / weight transposes
  cast_f32_bf16<<<(SEQ * DMODEL + TPB - 1) / TPB, TPB, 0, stream>>>(
      x, xb, SEQ * DMODEL);
  tcast_f32_bf16<<<(DMODEL * D3 + TPB - 1) / TPB, TPB, 0, stream>>>(
      w_qkv, wqkvT, DMODEL, D3);
  tcast_f32_bf16<<<(DMODEL * DMODEL + TPB - 1) / TPB, TPB, 0, stream>>>(
      w_out, woutT, DMODEL, DMODEL);

  // 2) QKV projection: [S,D] x [D,3D] + bias -> qkvb (bf16)
  gemm_bf16<false><<<dim3(D3 / 128, SEQ / 128), 128, 0, stream>>>(
      xb, wqkvT, b_qkv, (void*)qkvb, SEQ, D3, DMODEL);

  // 3) dual RoPE on q,k (in place)
  rope_kernel<<<(SEQ * DMODEL + TPB - 1) / TPB, TPB, 0, stream>>>(qkvb, posids);

  // 4) transpose V for contiguous PV B-fragments
  transpose_v<<<(SEQ * DMODEL + TPB - 1) / TPB, TPB, 0, stream>>>(qkvb, vtb);

  // 5) flash attention -> ctxb (bf16)
  flash_attn<<<dim3(SEQ / 64, NHEAD), 128, 0, stream>>>(qkvb, vtb, ctxb);

  // 6) output projection: [S,D] x [D,D] + bias -> d_out (f32)
  gemm_bf16<true><<<dim3(DMODEL / 128, SEQ / 128), 128, 0, stream>>>(
      ctxb, woutT, b_out, (void*)out, SEQ, DMODEL, DMODEL);
}
